// pqgnn_66494683676680
// MI455X (gfx1250) — matrix operand order repaired and checked by reading, not compiled, not run on hardware
//
#include <hip/hip_runtime.h>
#include <hip/hip_bf16.h>
#include <stddef.h>

// ---------------------------------------------------------------------------
// NRI-style GNN encoder for MI455X (gfx1250, wave32, WMMA f16 16x16x32).
// Adjacency is analytic (recv = e/95, send = e%95 (+1 if >= recv)), so
// node2edge = gather (fused into GEMM A-loads) and edge2node = scatter-add/95
// (fused into GEMM epilogue as global_atomic_add_f32).
//
// Big GEMMs: workgroup = 256 threads = 8 waves in 4(M)x2(N) grid; each wave
// owns a 32x64 output strip (2x4 grid of 16x16 WMMA tiles, 64 acc VGPRs).
// Weights (N x K f16) are staged once per workgroup into LDS; B fragments are
// ds_load_b128 pairs, A fragments are global_load_b128 pairs (gathered rows
// for the node2edge-fused variant).
// ---------------------------------------------------------------------------

typedef _Float16 half_t;
typedef _Float16 v16h __attribute__((ext_vector_type(16)));
typedef _Float16 v8h  __attribute__((ext_vector_type(8)));
typedef float    v8f  __attribute__((ext_vector_type(8)));

#define LN  96
#define BN  32
#define DIN 4
#define DN  128
#define NBN 3
#define NCN 6
#define EN  (LN * (LN - 1))   // 9120
#define BE  (BN * EN)         // 291840
#define BL  (BN * LN)         // 3072

__device__ __forceinline__ float elu_f(float x) {
    return x > 0.f ? x : (__expf(x) - 1.f);
}

// One v_wmma_f32_16x16x32_f16. Per-lane layouts (ISA 7.12.2):
//   A 16x32: row = lane&15, halves K = kb..kb+7 and 16+kb..16+kb+7, kb=(lane>>4)*8
//   B 32x16: col = lane&15, halves K = kb..kb+7 and 16+kb..16+kb+7, kb=(lane>>4)*8
__device__ __forceinline__ v8f wmma16(v8h alo, v8h ahi, v8h blo, v8h bhi, v8f c) {
    v16h a = __builtin_shufflevector(alo, ahi, 0,1,2,3,4,5,6,7,8,9,10,11,12,13,14,15);
    v16h b = __builtin_shufflevector(blo, bhi, 0,1,2,3,4,5,6,7,8,9,10,11,12,13,14,15);
    return __builtin_amdgcn_wmma_f32_16x16x32_f16(false, a, false, b, (short)0, c,
                                                  false, false);
}

// ---------------------------------------------------------------------------
// Weight prep: f32 (KxN) -> f16 transposed (NxK).
// ---------------------------------------------------------------------------
__global__ void k_wt_transpose(const float* __restrict__ W, half_t* __restrict__ Wt,
                               int K, int N) {
    int idx = blockIdx.x * 256 + threadIdx.x;
    if (idx >= K * N) return;
    int k = idx / N, n = idx - k * N;
    Wt[(size_t)n * K + k] = (half_t)W[idx];
}

// fW2 is 128x6; pad transposed copy to 16x128 (rows 6..15 zero).
__global__ void k_wt_transpose_fw2(const float* __restrict__ W, half_t* __restrict__ Wt) {
    int idx = blockIdx.x * 256 + threadIdx.x;   // 16*128
    if (idx >= 16 * DN) return;
    int n = idx / DN, k = idx - n * DN;
    Wt[idx] = (n < NCN) ? (half_t)W[k * NCN + n] : (half_t)0.f;
}

__global__ void k_zero_f32(float* __restrict__ p, int n) {
    int i = blockIdx.x * 256 + threadIdx.x;
    if (i < n) p[i] = 0.f;
}

__global__ void k_f32_to_f16(const float* __restrict__ s, half_t* __restrict__ d, int n) {
    int i = blockIdx.x * 256 + threadIdx.x;
    if (i < n) d[i] = (half_t)s[i];
}

// ---------------------------------------------------------------------------
// Input layer 1: t = ELU(x @ W0a + b0a), K = 4 (too small for WMMA; VALU).
// x[b,l,k] = inputs[l,b,k].  Writes f16 (BL x 128).
// ---------------------------------------------------------------------------
__global__ void k_input_l1(const float* __restrict__ X, const float* __restrict__ W0a,
                           const float* __restrict__ b0a, half_t* __restrict__ T) {
    int idx = blockIdx.x * 256 + threadIdx.x;       // BL*DN, exact
    int m = idx >> 7, col = idx & 127;
    int b = m / LN, l = m - b * LN;
    const float* xp = X + ((size_t)l * BN + b) * DIN;
    float v = b0a[col];
#pragma unroll
    for (int k = 0; k < DIN; ++k) v += xp[k] * W0a[k * DN + col];
    T[idx] = (half_t)elu_f(v);
}

// ---------------------------------------------------------------------------
// Workgroup-tile WMMA GEMM: 128(M) x 128(N) per block, K = KK (128 or 256).
//   GATHER : A rows come from node features via analytic node2edge
//            (K 0..127 = h[sender], K 128..255 = h[receiver]).
//   SCATTER: epilogue does ELU then atomic scatter-add/95 into NodeAcc
//            instead of storing the edge tensor.
// ---------------------------------------------------------------------------
template<int KK, bool GATHER, bool SCATTER>
__global__ __launch_bounds__(256)
void k_gemm_tile(const half_t* __restrict__ A,    // dense: M x KK ; gather: (B*L) x 128
                 const half_t* __restrict__ Wt,   // 128 x KK (N x K, f16)
                 const float* __restrict__ bias,  // 128
                 half_t* __restrict__ O,          // M x 128 f16 (when !SCATTER)
                 float* __restrict__ NodeAcc) {   // (B*L) x 128 f32 (when SCATTER)
    __shared__ half_t sw[128 * KK];               // 32 KB (K=128) / 64 KB (K=256)

    const int tid  = threadIdx.x;
    const int lane = tid & 31;
    const int wave = tid >> 5;
    const int wm   = wave >> 1;                   // 0..3 : M sub-block
    const int wn   = wave & 1;                    // 0..1 : N sub-block
    const int row0  = blockIdx.x * 128 + wm * 32;
    const int ncol0 = wn * 64;

    // --- stage weights into LDS (coalesced b128 copies) ---
    {
        const int total8 = 128 * KK / 8;
        for (int i = tid; i < total8; i += 256)
            *(v8h*)(sw + (size_t)i * 8) = *(const v8h*)(Wt + (size_t)i * 8);
    }
    __syncthreads();

    const int akb = (lane >> 4) * 8;
    const int bkb = (lane >> 4) * 8;

    // --- A row pointers for the wave's two M sub-tiles ---
    const half_t* ap[2];
    const half_t* hs[2];
    const half_t* hr[2];
    if constexpr (GATHER) {
#pragma unroll
        for (int mt = 0; mt < 2; ++mt) {
            int m    = row0 + mt * 16 + (lane & 15);
            int b    = m / EN;
            int e    = m - b * EN;
            int recv = e / (LN - 1);
            int rem  = e - recv * (LN - 1);
            int send = (rem < recv) ? rem : rem + 1;
            hs[mt] = A + (size_t)(b * LN + send) * DN;
            hr[mt] = A + (size_t)(b * LN + recv) * DN;
        }
    } else {
#pragma unroll
        for (int mt = 0; mt < 2; ++mt)
            ap[mt] = A + (size_t)(row0 + mt * 16 + (lane & 15)) * KK;
    }

    v8f acc[2][4] = {};

#pragma unroll
    for (int s = 0; s < KK / 32; ++s) {
        const int k0 = s * 32;
        v8h alo[2], ahi[2];
#pragma unroll
        for (int mt = 0; mt < 2; ++mt) {
            const half_t* src;
            int koff;
            if constexpr (GATHER) {
                src  = (s < 4) ? hs[mt] : hr[mt];
                koff = (s < 4) ? k0 : k0 - DN;
            } else {
                src  = ap[mt];
                koff = k0;
            }
            alo[mt] = *(const v8h*)(src + koff + akb);
            ahi[mt] = *(const v8h*)(src + koff + 16 + akb);
        }
#pragma unroll
        for (int nt = 0; nt < 4; ++nt) {
            const half_t* bp = sw + (size_t)(ncol0 + nt * 16 + (lane & 15)) * KK;
            v8h blo = *(const v8h*)(bp + k0 + bkb);       // ds_load_b128
            v8h bhi = *(const v8h*)(bp + k0 + 16 + bkb);  // ds_load_b128
#pragma unroll
            for (int mt = 0; mt < 2; ++mt)
                acc[mt][nt] = wmma16(alo[mt], ahi[mt], blo, bhi, acc[mt][nt]);
        }
    }

    // --- epilogue: bias + ELU, then store f16 or atomic scatter ---
#pragma unroll
    for (int nt = 0; nt < 4; ++nt) {
        const int col = ncol0 + nt * 16 + (lane & 15);
        const float bs = bias[col];
#pragma unroll
        for (int mt = 0; mt < 2; ++mt) {
            const int rbase = row0 + mt * 16 + ((lane >> 4) * 8);
#pragma unroll
            for (int r = 0; r < 8; ++r) {
                float v = elu_f(acc[mt][nt][r] + bs);
                if constexpr (SCATTER) {
                    int m    = rbase + r;
                    int b    = m / EN;
                    int e    = m - b * EN;
                    int recv = e / (LN - 1);
                    atomicAdd(&NodeAcc[(size_t)(b * LN + recv) * DN + col],
                              v * (1.f / (float)(LN - 1)));
                } else {
                    O[(size_t)(rbase + r) * DN + col] = (half_t)v;
                }
            }
        }
    }
}

// ---------------------------------------------------------------------------
// Final GEMM2: (B*E) x 128 @ 128 x 6 (padded to 16) + bias, f32 out, no ELU.
// One wave per 16-row tile.
// ---------------------------------------------------------------------------
__global__ void k_final_gemm2(const half_t* __restrict__ A,   // (B*E) x 128
                              const half_t* __restrict__ Wt,  // 16 x 128 (padded)
                              const float* __restrict__ bias, // 6
                              float* __restrict__ Out) {      // (B*E) x 6
    const int lane = threadIdx.x & 31;
    const int row0 = blockIdx.x * 16;
    const int akb  = (lane >> 4) * 8;
    const int bcol = lane & 15;
    const int bkb  = (lane >> 4) * 8;
    const half_t* ap = A + (size_t)(row0 + (lane & 15)) * DN;
    const half_t* bp = Wt + (size_t)bcol * DN;
    v8f acc = {};
#pragma unroll
    for (int k0 = 0; k0 < DN; k0 += 32) {
        v8h alo = *(const v8h*)(ap + k0 + akb);
        v8h ahi = *(const v8h*)(ap + k0 + 16 + akb);
        v8h blo = *(const v8h*)(bp + k0 + bkb);
        v8h bhi = *(const v8h*)(bp + k0 + 16 + bkb);
        acc = wmma16(alo, ahi, blo, bhi, acc);
    }
    if (bcol < NCN) {
        const float bs = bias[bcol];
        const int rbase = row0 + ((lane >> 4) * 8);
#pragma unroll
        for (int r = 0; r < 8; ++r)
            Out[(size_t)(rbase + r) * NCN + bcol] = acc[r] + bs;
    }
}

// ---------------------------------------------------------------------------
// Host-side pipeline (all on `stream`; graph-capture safe: no malloc/sync).
// ---------------------------------------------------------------------------
extern "C" void kernel_launch(void* const* d_in, const int* in_sizes, int n_in,
                              void* d_out, int out_size, void* d_ws, size_t ws_size,
                              hipStream_t stream) {
    (void)in_sizes; (void)n_in; (void)out_size; (void)ws_size;

    const float* inputs = (const float*)d_in[0];
    // d_in[1], d_in[2] (rel_rec / rel_send) are analytic one-hots: unused.
    const float* W0a = (const float*)d_in[3];
    const float* b0a = (const float*)d_in[4];
    const float* W0b = (const float*)d_in[5];
    const float* b0b = (const float*)d_in[6];
    const float* eW1 = (const float*)d_in[7];
    const float* eb1 = (const float*)d_in[8];
    const float* eW2 = (const float*)d_in[9];
    const float* eb2 = (const float*)d_in[10];
    const float* nW1 = (const float*)d_in[11];
    const float* nb1 = (const float*)d_in[12];
    const float* nW2 = (const float*)d_in[13];
    const float* nb2 = (const float*)d_in[14];
    const float* fW1 = (const float*)d_in[15];
    const float* fb1 = (const float*)d_in[16];
    const float* fW2 = (const float*)d_in[17];
    const float* fb2 = (const float*)d_in[18];
    float* out = (float*)d_out;

    char* ws = (char*)d_ws;
    size_t cur = 0;
    auto alloc = [&](size_t bytes) -> char* {
        char* p = ws + cur;
        cur = (cur + bytes + 255) & ~(size_t)255;
        return p;
    };

    half_t* wt_w0b = (half_t*)alloc((size_t)DN * DN * 2);
    half_t* wt_e1  = (half_t*)alloc((size_t)NBN * 2 * DN * DN * 2);
    half_t* wt_e2  = (half_t*)alloc((size_t)NBN * DN * DN * 2);
    half_t* wt_n1  = (half_t*)alloc((size_t)NBN * DN * DN * 2);
    half_t* wt_n2  = (half_t*)alloc((size_t)NBN * DN * DN * 2);
    half_t* wt_f1  = (half_t*)alloc((size_t)2 * DN * DN * 2);
    half_t* wt_f2  = (half_t*)alloc((size_t)16 * DN * 2);
    half_t* h      = (half_t*)alloc((size_t)BL * DN * 2);
    half_t* tbuf   = (half_t*)alloc((size_t)BL * DN * 2);
    half_t* nodex  = (half_t*)alloc((size_t)BL * DN * 2);
    float*  nacc   = (float*) alloc((size_t)BL * DN * 4);
    half_t* ehid   = (half_t*)alloc((size_t)BE * DN * 2);   // 75 MB edge hidden

    auto g = [](int n) { return (n + 255) / 256; };

    // --- weight conversion / transposition (f32 KxN -> f16 NxK) ---
    k_wt_transpose<<<g(DN * DN), 256, 0, stream>>>(W0b, wt_w0b, DN, DN);
    for (int i = 0; i < NBN; ++i) {
        k_wt_transpose<<<g(2 * DN * DN), 256, 0, stream>>>(
            eW1 + (size_t)i * 2 * DN * DN, wt_e1 + (size_t)i * 2 * DN * DN, 2 * DN, DN);
        k_wt_transpose<<<g(DN * DN), 256, 0, stream>>>(
            eW2 + (size_t)i * DN * DN, wt_e2 + (size_t)i * DN * DN, DN, DN);
        k_wt_transpose<<<g(DN * DN), 256, 0, stream>>>(
            nW1 + (size_t)i * DN * DN, wt_n1 + (size_t)i * DN * DN, DN, DN);
        k_wt_transpose<<<g(DN * DN), 256, 0, stream>>>(
            nW2 + (size_t)i * DN * DN, wt_n2 + (size_t)i * DN * DN, DN, DN);
    }
    k_wt_transpose<<<g(2 * DN * DN), 256, 0, stream>>>(fW1, wt_f1, 2 * DN, DN);
    k_wt_transpose_fw2<<<g(16 * DN), 256, 0, stream>>>(fW2, wt_f2);

    // --- input MLP: h = ELU(ELU(x@W0a+b0a)@W0b+b0b) ---
    k_input_l1<<<BL * DN / 256, 256, 0, stream>>>(inputs, W0a, b0a, tbuf);
    k_gemm_tile<DN, false, false><<<BL / 128, 256, 0, stream>>>(
        tbuf, wt_w0b, b0b, h, nullptr);

    // --- message-passing blocks ---
    for (int i = 0; i < NBN; ++i) {
        k_gemm_tile<2 * DN, true, false><<<BE / 128, 256, 0, stream>>>(
            h, wt_e1 + (size_t)i * 2 * DN * DN, eb1 + i * DN, ehid, nullptr);
        k_zero_f32<<<g(BL * DN), 256, 0, stream>>>(nacc, BL * DN);
        k_gemm_tile<DN, false, true><<<BE / 128, 256, 0, stream>>>(
            ehid, wt_e2 + (size_t)i * DN * DN, eb2 + i * DN, nullptr, nacc);
        k_f32_to_f16<<<g(BL * DN), 256, 0, stream>>>(nacc, nodex, BL * DN);
        k_gemm_tile<DN, false, false><<<BL / 128, 256, 0, stream>>>(
            nodex, wt_n1 + (size_t)i * DN * DN, nb1 + i * DN, tbuf, nullptr);
        k_gemm_tile<DN, false, false><<<BL / 128, 256, 0, stream>>>(
            tbuf, wt_n2 + (size_t)i * DN * DN, nb2 + i * DN, h, nullptr);
    }

    // --- final readout: ELU(node2edge(h)@fW1+fb1)@fW2+fb2 ---
    k_gemm_tile<2 * DN, true, false><<<BE / 128, 256, 0, stream>>>(
        h, wt_f1, fb1, ehid, nullptr);
    k_final_gemm2<<<BE / 16, 32, 0, stream>>>(ehid, wt_f2, fb2, out);
}
